// LSTM_Seq2seq_82669530514119
// MI455X (gfx1250) — matrix operand order repaired and checked by reading, not compiled, not run on hardware
//
#include <hip/hip_runtime.h>
#include <hip/hip_bf16.h>

typedef __bf16 bf16_t;
typedef __attribute__((ext_vector_type(16))) __bf16 v16bf;
typedef __attribute__((ext_vector_type(8)))  __bf16 v8bf;
typedef __attribute__((ext_vector_type(4)))  __bf16 v4bf;
typedef __attribute__((ext_vector_type(8)))  float  v8f;

#define BSZ   256
#define T_IN  256
#define T_OUT 256
#define D_IN  512
#define HIDN  1024
#define OUTD  1024

__device__ __forceinline__ float sigmoidf_(float x) {
    return 1.0f / (1.0f + __expf(-x));
}

// ---------------------------------------------------------------------------
// Pack a row-major f32 weight matrix [Kdim x Nfull] into bf16 WMMA B-fragment
// order for v_wmma_f32_16x16x32_bf16:
//   frag(nt, kt) is 512 contiguous bf16; element e (lane l = e>>4, half h = e&15):
//     K = kt*32 + (h>>3)*16 + (l>>4)*8 + (h&7),  N = nt*16 + (l&15)
// so in the GEMM a lane reads 16 contiguous bf16 (32B) at base + lane*16.
// ---------------------------------------------------------------------------
__global__ __launch_bounds__(256) void pack_w(const float* __restrict__ W,
                                              bf16_t* __restrict__ out,
                                              int Kdim, int Nfull) {
    size_t idx = (size_t)blockIdx.x * 256 + threadIdx.x;
    size_t total = (size_t)Kdim * Nfull;
    if (idx >= total) return;
    int KT = Kdim >> 5;
    size_t frag = idx >> 9;
    int e  = (int)(idx & 511);
    int nt = (int)(frag / KT);
    int kt = (int)(frag % KT);
    int l  = e >> 4;
    int h  = e & 15;
    int K  = kt * 32 + (h >> 3) * 16 + (l >> 4) * 8 + (h & 7);
    int N  = nt * 16 + (l & 15);
    out[idx] = (bf16_t)W[(size_t)K * Nfull + N];
}

// ---------------------------------------------------------------------------
// Fused LSTM step: gates = [x_t, h_prev] @ W + b ; c,h update ; optional output.
// Block: 256 threads (8 waves). Grid: (BSZ/32, NH/16/8).
// Block stages a 32-row activation panel; each wave computes TWO 16x16
// (batch x hidden) patches for all 4 gates (8 accumulators), so every B
// fragment load feeds two WMMAs.
// ---------------------------------------------------------------------------
template <int KX, int KH, bool XF32>
__global__ __launch_bounds__(256) void lstm_step(
    const void*  __restrict__ xsrc, size_t x_row_stride, size_t x_off,
    const bf16_t* __restrict__ h_prev,
    float*        __restrict__ c_state,
    bf16_t*       __restrict__ h_next,
    const bf16_t* __restrict__ Wp,
    const float*  __restrict__ bias,
    float*        __restrict__ outp, size_t out_row_stride) {

    constexpr int KP = KX + KH;     // panel width
    constexpr int KT = KP / 32;     // k-steps
    __shared__ bf16_t panel[32 * KP];

    const int tid = threadIdx.x;
    const int mb  = blockIdx.x;     // rows [mb*32, mb*32+32)

    // ---- stage A panel (32 rows x KP cols, bf16) into LDS ----
    if constexpr (XF32) {
        const float* xf = (const float*)xsrc;
        for (int i = tid; i < 32 * (KX / 4); i += 256) {
            int r = i / (KX / 4);
            int c = (i % (KX / 4)) * 4;
            const float4 v = *(const float4*)(xf + (size_t)(mb * 32 + r) * x_row_stride + x_off + c);
            v4bf o;
            o[0] = (bf16_t)v.x; o[1] = (bf16_t)v.y;
            o[2] = (bf16_t)v.z; o[3] = (bf16_t)v.w;
            *(v4bf*)&panel[(size_t)r * KP + c] = o;
        }
    } else {
        const bf16_t* xb = (const bf16_t*)xsrc;
        for (int i = tid; i < 32 * (KX / 8); i += 256) {
            int r = i / (KX / 8);
            int c = (i % (KX / 8)) * 8;
            *(v8bf*)&panel[(size_t)r * KP + c] =
                *(const v8bf*)(xb + (size_t)(mb * 32 + r) * x_row_stride + x_off + c);
        }
    }
    for (int i = tid; i < 32 * (KH / 8); i += 256) {
        int r = i / (KH / 8);
        int c = (i % (KH / 8)) * 8;
        *(v8bf*)&panel[(size_t)r * KP + KX + c] =
            *(const v8bf*)(h_prev + (size_t)(mb * 32 + r) * KH + c);
    }
    __syncthreads();

    const int wave = tid >> 5;
    const int lane = tid & 31;
    const int jt   = blockIdx.y * 8 + wave;   // hidden n-tile: 0..KH/16-1
    const int r    = lane & 15;
    const int hi   = lane >> 4;

    v8f aA0 = {0,0,0,0,0,0,0,0}, aA1 = {0,0,0,0,0,0,0,0};
    v8f aA2 = {0,0,0,0,0,0,0,0}, aA3 = {0,0,0,0,0,0,0,0};
    v8f aB0 = {0,0,0,0,0,0,0,0}, aB1 = {0,0,0,0,0,0,0,0};
    v8f aB2 = {0,0,0,0,0,0,0,0}, aB3 = {0,0,0,0,0,0,0,0};

    constexpr int NTG = KH / 16;   // n-tiles per gate
    const bf16_t* w0 = Wp + ((size_t)(0 * NTG + jt) * KT) * 512 + (size_t)lane * 16;
    const bf16_t* w1 = Wp + ((size_t)(1 * NTG + jt) * KT) * 512 + (size_t)lane * 16;
    const bf16_t* w2 = Wp + ((size_t)(2 * NTG + jt) * KT) * 512 + (size_t)lane * 16;
    const bf16_t* w3 = Wp + ((size_t)(3 * NTG + jt) * KT) * 512 + (size_t)lane * 16;

    for (int kt = 0; kt < KT; ++kt) {
        // B fragments for the 4 gates (each feeds two WMMAs)
        v16bf b0 = *(const v16bf*)(w0 + (size_t)kt * 512);
        v16bf b1 = *(const v16bf*)(w1 + (size_t)kt * 512);
        v16bf b2 = *(const v16bf*)(w2 + (size_t)kt * 512);
        v16bf b3 = *(const v16bf*)(w3 + (size_t)kt * 512);

        // A fragments for the two 16-row halves
        const bf16_t* paA = &panel[(size_t)r * KP + kt * 32 + hi * 8];
        v8bf loA = *(const v8bf*)paA;
        v8bf hiA = *(const v8bf*)(paA + 16);
        const bf16_t* paB = &panel[(size_t)(r + 16) * KP + kt * 32 + hi * 8];
        v8bf loB = *(const v8bf*)paB;
        v8bf hiB = *(const v8bf*)(paB + 16);
        v16bf afA, afB;
#pragma unroll
        for (int q = 0; q < 8; ++q) {
            afA[q] = loA[q]; afA[q + 8] = hiA[q];
            afB[q] = loB[q]; afB[q + 8] = hiB[q];
        }

        aA0 = __builtin_amdgcn_wmma_f32_16x16x32_bf16(false, afA, false, b0, (short)0, aA0, false, false);
        aB0 = __builtin_amdgcn_wmma_f32_16x16x32_bf16(false, afB, false, b0, (short)0, aB0, false, false);
        aA1 = __builtin_amdgcn_wmma_f32_16x16x32_bf16(false, afA, false, b1, (short)0, aA1, false, false);
        aB1 = __builtin_amdgcn_wmma_f32_16x16x32_bf16(false, afB, false, b1, (short)0, aB1, false, false);
        aA2 = __builtin_amdgcn_wmma_f32_16x16x32_bf16(false, afA, false, b2, (short)0, aA2, false, false);
        aB2 = __builtin_amdgcn_wmma_f32_16x16x32_bf16(false, afB, false, b2, (short)0, aB2, false, false);
        aA3 = __builtin_amdgcn_wmma_f32_16x16x32_bf16(false, afA, false, b3, (short)0, aA3, false, false);
        aB3 = __builtin_amdgcn_wmma_f32_16x16x32_bf16(false, afB, false, b3, (short)0, aB3, false, false);
    }

    // ---- epilogue: activations + state update ----
    const int ncol = jt * 16 + (lane & 15);
    const float bi = bias[0 * KH + ncol];
    const float bf = bias[1 * KH + ncol];
    const float bo = bias[2 * KH + ncol];
    const float bg = bias[3 * KH + ncol];

#pragma unroll
    for (int half = 0; half < 2; ++half) {
        v8f g0 = half ? aB0 : aA0;
        v8f g1 = half ? aB1 : aA1;
        v8f g2 = half ? aB2 : aA2;
        v8f g3 = half ? aB3 : aA3;
#pragma unroll
        for (int v = 0; v < 8; ++v) {
            int row = mb * 32 + half * 16 + hi * 8 + v;   // D layout: M = v + hi*8
            size_t idx = (size_t)row * KH + ncol;
            float iv = sigmoidf_(g0[v] + bi);
            float fv = sigmoidf_(g1[v] + bf);
            float ov = sigmoidf_(g2[v] + bo);
            float gv = tanhf(g3[v] + bg);
            float cn = fv * c_state[idx] + iv * gv;
            c_state[idx] = cn;
            float hn = ov * tanhf(cn);
            h_next[idx] = (bf16_t)hn;
            if (outp) outp[(size_t)row * out_row_stride + ncol] = hn;
        }
    }
}

// ---------------------------------------------------------------------------
// Projection GEMM: out = src[256x1024] @ Wp + bias, out to f32 and/or bf16.
// ---------------------------------------------------------------------------
template <bool XF32>
__global__ __launch_bounds__(256) void proj_gemm(
    const void*   __restrict__ src,
    const bf16_t* __restrict__ Wp,
    const float*  __restrict__ bias,
    float*        __restrict__ outF,
    bf16_t*       __restrict__ outB) {

    constexpr int K  = 1024;
    constexpr int KT = K / 32;
    __shared__ bf16_t panel[16 * K];

    const int tid = threadIdx.x;
    const int mt  = blockIdx.x;

    if constexpr (XF32) {
        const float* xf = (const float*)src;
        for (int i = tid; i < 16 * (K / 4); i += 256) {
            int r = i / (K / 4);
            int c = (i % (K / 4)) * 4;
            const float4 v = *(const float4*)(xf + (size_t)(mt * 16 + r) * K + c);
            v4bf o;
            o[0] = (bf16_t)v.x; o[1] = (bf16_t)v.y;
            o[2] = (bf16_t)v.z; o[3] = (bf16_t)v.w;
            *(v4bf*)&panel[(size_t)r * K + c] = o;
        }
    } else {
        const bf16_t* xb = (const bf16_t*)src;
        for (int i = tid; i < 16 * (K / 8); i += 256) {
            int r = i / (K / 8);
            int c = (i % (K / 8)) * 8;
            *(v8bf*)&panel[(size_t)r * K + c] =
                *(const v8bf*)(xb + (size_t)(mt * 16 + r) * K + c);
        }
    }
    __syncthreads();

    const int wave = tid >> 5;
    const int lane = tid & 31;
    const int jt   = blockIdx.y * 8 + wave;
    const int r    = lane & 15;
    const int hi   = lane >> 4;

    v8f acc = {0,0,0,0,0,0,0,0};
    const bf16_t* w = Wp + ((size_t)jt * KT) * 512 + (size_t)lane * 16;

    for (int kt = 0; kt < KT; ++kt) {
        v16bf bfr = *(const v16bf*)(w + (size_t)kt * 512);
        const bf16_t* pa = &panel[(size_t)r * K + kt * 32 + hi * 8];
        v8bf alo = *(const v8bf*)pa;
        v8bf ahi = *(const v8bf*)(pa + 16);
        v16bf af;
#pragma unroll
        for (int q = 0; q < 8; ++q) { af[q] = alo[q]; af[q + 8] = ahi[q]; }
        acc = __builtin_amdgcn_wmma_f32_16x16x32_bf16(false, af, false, bfr, (short)0, acc, false, false);
    }

    const int ncol = jt * 16 + (lane & 15);
    const float bb = bias[ncol];
#pragma unroll
    for (int v = 0; v < 8; ++v) {
        int row = mt * 16 + hi * 8 + v;
        size_t idx = (size_t)row * 1024 + ncol;
        float val = acc[v] + bb;
        if (outF) outF[idx] = val;
        if (outB) outB[idx] = (bf16_t)val;
    }
}

// ---------------------------------------------------------------------------
extern "C" void kernel_launch(void* const* d_in, const int* in_sizes, int n_in,
                              void* d_out, int out_size, void* d_ws, size_t ws_size,
                              hipStream_t stream) {
    (void)in_sizes; (void)n_in; (void)out_size; (void)ws_size;

    const float* x     = (const float*)d_in[0];
    const float* enc_W = (const float*)d_in[1];
    const float* enc_b = (const float*)d_in[2];
    const float* dec_W = (const float*)d_in[3];
    const float* dec_b = (const float*)d_in[4];
    const float* Wh    = (const float*)d_in[5];
    const float* bh    = (const float*)d_in[6];
    const float* Wc    = (const float*)d_in[7];
    const float* bc    = (const float*)d_in[8];
    float* out = (float*)d_out;

    char* ws = (char*)d_ws;
    auto alloc = [&](size_t bytes) -> char* {
        char* p = ws;
        ws += (bytes + 255) & ~(size_t)255;
        return p;
    };
    bf16_t* encp = (bf16_t*)alloc((size_t)1536 * 4096 * 2);
    bf16_t* decp = (bf16_t*)alloc((size_t)2048 * 4096 * 2);
    bf16_t* whp  = (bf16_t*)alloc((size_t)1024 * 1024 * 2);
    bf16_t* wcp  = (bf16_t*)alloc((size_t)1024 * 1024 * 2);
    bf16_t* eh0  = (bf16_t*)alloc((size_t)BSZ * HIDN * 2);
    bf16_t* eh1  = (bf16_t*)alloc((size_t)BSZ * HIDN * 2);
    float*  ec   = (float*) alloc((size_t)BSZ * HIDN * 4);
    bf16_t* dh0  = (bf16_t*)alloc((size_t)BSZ * OUTD * 2);
    bf16_t* dh1  = (bf16_t*)alloc((size_t)BSZ * OUTD * 2);
    float*  dc   = (float*) alloc((size_t)BSZ * OUTD * 4);

    // One-time (per launch) weight repack f32 -> bf16 fragment order.
    pack_w<<<(unsigned)(((size_t)1536 * 4096 + 255) / 256), 256, 0, stream>>>(enc_W, encp, 1536, 4096);
    pack_w<<<(unsigned)(((size_t)2048 * 4096 + 255) / 256), 256, 0, stream>>>(dec_W, decp, 2048, 4096);
    pack_w<<<(unsigned)(((size_t)1024 * 1024 + 255) / 256), 256, 0, stream>>>(Wh, whp, 1024, 1024);
    pack_w<<<(unsigned)(((size_t)1024 * 1024 + 255) / 256), 256, 0, stream>>>(Wc, wcp, 1024, 1024);

    hipMemsetAsync(eh0, 0, (size_t)BSZ * HIDN * 2, stream);
    hipMemsetAsync(ec,  0, (size_t)BSZ * HIDN * 4, stream);

    // ---- encoder scan ----
    for (int t = 0; t < T_IN; ++t) {
        bf16_t* hp = (t & 1) ? eh1 : eh0;
        bf16_t* hn = (t & 1) ? eh0 : eh1;
        lstm_step<D_IN, HIDN, true><<<dim3(BSZ / 32, HIDN / 16 / 8), 256, 0, stream>>>(
            x, (size_t)T_IN * D_IN, (size_t)t * D_IN,
            hp, ec, hn, encp, enc_b, nullptr, 0);
    }
    bf16_t* h_enc = eh0;  // final state after 256 steps (even count)

    // ---- decoder initial state projections ----
    proj_gemm<false><<<dim3(BSZ / 16, 8), 256, 0, stream>>>(h_enc, whp, bh, nullptr, dh0);
    proj_gemm<true ><<<dim3(BSZ / 16, 8), 256, 0, stream>>>(ec,    wcp, bc, dc, nullptr);

    // ---- decoder scan (constant input h_enc), emit h each step ----
    for (int t = 0; t < T_OUT; ++t) {
        bf16_t* hp = (t & 1) ? dh1 : dh0;
        bf16_t* hn = (t & 1) ? dh0 : dh1;
        lstm_step<HIDN, OUTD, false><<<dim3(BSZ / 32, OUTD / 16 / 8), 256, 0, stream>>>(
            h_enc, (size_t)HIDN, 0,
            hp, dc, hn, decp, dec_b,
            out + (size_t)t * OUTD, (size_t)T_OUT * OUTD);
    }
}
